// CompressiveEncoder_171798692210
// MI455X (gfx1250) — matrix operand order repaired and checked by reading, not compiled, not run on hardware
//
#include <hip/hip_runtime.h>

#define TT 4
#define NLAYERS 4
#define BB 8
#define SS 512
#define DD 512
#define HH 8
#define DHH 64
#define FFF 2048
#define MK 1536
#define VOC 1000
#define SCALE_F 0.35355339059327373f   // 1/sqrt(H)
#define NEG_F (-1.0e9f)

typedef __attribute__((ext_vector_type(16))) _Float16 v16h;
typedef __attribute__((ext_vector_type(8)))  float    v8f;
typedef unsigned int u32x4 __attribute__((ext_vector_type(4)));
typedef int          i32x4 __attribute__((ext_vector_type(4)));
typedef int          i32x8 __attribute__((ext_vector_type(8)));

union Frag16 { unsigned int u[8]; v16h v; };

// ---------------------------------------------------------------------------
// TDM: DMA one 64-row x 32-col f16 tile (row stride ldElems) into LDS at
// ldsAddr, hardware-padding each 64B row with 8B so the LDS row stride is
// 36 halves (matches the WMMA fragment loads). D# per ISA 08 §8.3/8.4.
// ---------------------------------------------------------------------------
static __device__ __forceinline__ void tdm_load_tile_f16(
    const _Float16* gsrc, unsigned ldsAddr, long ldElems)
{
    unsigned long long ga = (unsigned long long)(const void*)gsrc;
    u32x4 g0;
    g0[0] = 1u;                                   // count=1, user mode
    g0[1] = ldsAddr;                              // LDS byte address
    g0[2] = (unsigned)ga;                         // global_addr[31:0]
    g0[3] = ((unsigned)(ga >> 32) & 0x1FFFFFFu)   // global_addr[56:32]
          | (2u << 30);                           // type=2 ("image")
    i32x8 g1;
    g1[0] = (1 << 16)      // data_size = 2 bytes
          | (1 << 20)      // pad_enable
          | (3 << 22)      // pad_interval: 16 DWORDs (=64B row)
          | (1 << 25);     // pad_amount:   2 DWORDs (=8B)
    g1[1] = 0;                                        // td0[15:0]=0 (td0=1<<30)
    g1[2] = 0x4000;                                   // td0[31:16]; td1[15:0]=0
    g1[3] = 0x4000 | (32 << 16);                      // td1[31:16]; tile_dim0=32
    g1[4] = 64;                                       // tile_dim1=64, tile_dim2=0
    g1[5] = (int)((unsigned long long)ldElems & 0xFFFFFFFFull);   // stride0 lo
    g1[6] = (int)(((unsigned long long)ldElems >> 32) & 0xFFFFull); // stride0 hi
    g1[7] = 0;
    i32x4 gz; gz[0] = 0; gz[1] = 0; gz[2] = 0; gz[3] = 0;
#if defined(__clang_major__) && (__clang_major__ >= 23)
    i32x8 gz8; gz8[0]=0; gz8[1]=0; gz8[2]=0; gz8[3]=0; gz8[4]=0; gz8[5]=0; gz8[6]=0; gz8[7]=0;
    __builtin_amdgcn_tensor_load_to_lds(g0, g1, gz, gz, gz8, 0);
#else
    __builtin_amdgcn_tensor_load_to_lds(g0, g1, gz, gz, 0);
#endif
}

// ---------------------------------------------------------------------------
// All-f16 batched WMMA GEMM:  C = act(A @ B^T + bias), both operands f16.
//   A: rows m (M), cols k (K), row stride lda          (f16)
//   B: rows n (N), cols k (K), row stride ldb          (f16, "(N,K)" layout)
//   batch z: zo=z/inner, zi=z%inner; ptr += zo*off1 + zi*off2
//   64x64 tile per block, BK=32, 128 thr = 4 wave32 waves (each 32x32).
//   Tiles double-buffered in LDS, fetched by TDM (wave0: A, wave1: B),
//   ordered with s_wait_tensorcnt + barrier.  Epilogue -> f32 and/or f16.
// ---------------------------------------------------------------------------
__global__ __launch_bounds__(128) void wmma_gemm_f16(
    const _Float16* __restrict__ A, long offA1, long offA2, int lda,
    const _Float16* __restrict__ B, long offB1, long offB2, int ldb,
    float* __restrict__ Cf, _Float16* __restrict__ Ch,
    long offC1, long offC2, int ldc,
    int K, int inner,
    const float* __restrict__ bias, int act)
{
    int z  = blockIdx.z;
    int zo = z / inner;
    int zi = z - zo * inner;
    const int n0  = blockIdx.x * 64;
    const int m0  = blockIdx.y * 64;
    A += zo * offA1 + zi * offA2 + (long)m0 * lda;
    B += zo * offB1 + zi * offB2 + (long)n0 * ldb;
    Cf = Cf ? Cf + zo * offC1 + zi * offC2 : Cf;
    Ch = Ch ? Ch + zo * offC1 + zi * offC2 : Ch;

    const int tid  = threadIdx.x;
    const int lane = tid & 31;
    const int wave = tid >> 5;          // 0..3
    const int wm   = wave >> 1;
    const int wn   = wave & 1;
    const int l15  = lane & 15;
    const int kh   = lane >> 4;

    // [buf][0]=A tile, [buf][1]=B tile; 64 rows x 36 halves (TDM-padded)
    __shared__ _Float16 tiles[2][2][64][36];
    const unsigned tileBytes = 64 * 36 * 2;
    const unsigned ldsBase = (unsigned)(unsigned long long)(const void*)&tiles[0][0][0][0];

    v8f zero8 = {0.f,0.f,0.f,0.f,0.f,0.f,0.f,0.f};
    v8f acc[2][2] = {{zero8, zero8}, {zero8, zero8}};

    auto issue = [&](int kb, int buf) {
        if (wave == 0)
            tdm_load_tile_f16(A + ((long)kb << 5), ldsBase + (unsigned)buf * 2u * tileBytes, lda);
        else if (wave == 1)
            tdm_load_tile_f16(B + ((long)kb << 5), ldsBase + (unsigned)buf * 2u * tileBytes + tileBytes, ldb);
    };
    auto compute = [&](int cur) {
        const _Float16 (*As)[36] = tiles[cur][0];
        const _Float16 (*Bs)[36] = tiles[cur][1];
        Frag16 af[2], bf[2];
        #pragma unroll
        for (int mi = 0; mi < 2; ++mi) {
            int row = wm*32 + mi*16 + l15;
            #pragma unroll
            for (int j = 0; j < 8; ++j) {
                int kj = (j < 4) ? (kh*8 + 2*j) : (16 + kh*8 + 2*(j-4));
                af[mi].u[j] = *(const unsigned int*)&As[row][kj];
            }
        }
        #pragma unroll
        for (int ni = 0; ni < 2; ++ni) {
            int col = wn*32 + ni*16 + l15;
            #pragma unroll
            for (int j = 0; j < 8; ++j)
                bf[ni].u[j] = *(const unsigned int*)&Bs[col][(j < 4) ? (kh*8 + 2*j) : (16 + kh*8 + 2*(j-4))];
        }
        #pragma unroll
        for (int mi = 0; mi < 2; ++mi)
            #pragma unroll
            for (int ni = 0; ni < 2; ++ni)
                acc[mi][ni] = __builtin_amdgcn_wmma_f32_16x16x32_f16(
                    false, af[mi].v, false, bf[ni].v,
                    (short)0, acc[mi][ni], false, false);
    };

    const int nK = K >> 5;
    issue(0, 0);
    for (int kb = 0; kb < nK - 1; ++kb) {
        int cur = kb & 1;
        issue(kb + 1, cur ^ 1);                 // prefetch next slab via TDM
        __builtin_amdgcn_s_wait_tensorcnt(1);   // slab kb has landed
        __syncthreads();
        compute(cur);
        __syncthreads();                        // done reading buf[cur]
    }
    __builtin_amdgcn_s_wait_tensorcnt(0);
    __syncthreads();
    compute((nK - 1) & 1);

    // Epilogue: C/D layout — VGPR j: M = j + 8*kh, N = l15.
    #pragma unroll
    for (int mi = 0; mi < 2; ++mi) {
        #pragma unroll
        for (int ni = 0; ni < 2; ++ni) {
            int col = n0 + wn*32 + ni*16 + l15;
            float bv = bias ? bias[col] : 0.0f;
            #pragma unroll
            for (int j = 0; j < 8; ++j) {
                int row = m0 + wm*32 + mi*16 + j + kh*8;
                float val = acc[mi][ni][j] + bv;
                if (act) val = 0.5f * val * (1.0f + erff(val * 0.70710678118654752f));
                long idx = (long)row * ldc + col;
                if (Cf) Cf[idx] = val;
                if (Ch) Ch[idx] = (_Float16)val;
            }
        }
    }
}

// ---------------------------------------------------------------------------
// Weight prep: dst[c*dstStride + r] = (f16) src[r*srcStride + c]  (batched)
// ---------------------------------------------------------------------------
__global__ __launch_bounds__(256) void trans_f32_f16(
    const float* __restrict__ src, long sOff, _Float16* __restrict__ dst, long dOff,
    int srcStride, int dstStride)
{
    __shared__ float tile[32][33];
    src += (long)blockIdx.z * sOff;
    dst += (long)blockIdx.z * dOff;
    int c0 = blockIdx.x * 32, r0 = blockIdx.y * 32;
    int tx = threadIdx.x & 31, ty = threadIdx.x >> 5;
    #pragma unroll
    for (int i = 0; i < 32; i += 8)
        tile[ty + i][tx] = src[(long)(r0 + ty + i) * srcStride + (c0 + tx)];
    __syncthreads();
    #pragma unroll
    for (int i = 0; i < 32; i += 8)
        dst[(long)(c0 + ty + i) * dstStride + (r0 + tx)] = (_Float16)tile[tx][ty + i];
}

// f16 -> f16 transpose with two-level batch offset (per-(b,h) V transpose)
__global__ __launch_bounds__(256) void trans_f16_f16(
    const _Float16* __restrict__ src, long sOff1, long sOff2, int inner,
    _Float16* __restrict__ dst, long dOff, int srcStride, int dstStride)
{
    __shared__ _Float16 tile[32][33];
    int z = blockIdx.z, zo = z / inner, zi = z - zo * inner;
    src += zo * sOff1 + zi * sOff2;
    dst += (long)z * dOff;
    int c0 = blockIdx.x * 32, r0 = blockIdx.y * 32;
    int tx = threadIdx.x & 31, ty = threadIdx.x >> 5;
    #pragma unroll
    for (int i = 0; i < 32; i += 8)
        tile[ty + i][tx] = src[(long)(r0 + ty + i) * srcStride + (c0 + tx)];
    __syncthreads();
    #pragma unroll
    for (int i = 0; i < 32; i += 8)
        dst[(long)(c0 + ty + i) * dstStride + (r0 + tx)] = tile[tx][ty + i];
}

// r_emb window (last S rows) f32 -> f16; grid (S*DH/256, T*N*H)
__global__ __launch_bounds__(256) void conv_re(
    const float* __restrict__ src, _Float16* __restrict__ dst)
{
    long si = (long)blockIdx.y * MK * DHH + (long)(MK - SS) * DHH;
    long di = (long)blockIdx.y * SS * DHH;
    int i = blockIdx.x * 256 + threadIdx.x;
    dst[di + i] = (_Float16)src[si + i];
}

// x[row,:] = embed[seq[row], :]  (+ f16 mirror)
__global__ __launch_bounds__(256) void embed_gather(
    const int* __restrict__ seq, const float* __restrict__ emb,
    float* __restrict__ x, _Float16* __restrict__ xh)
{
    int row = blockIdx.x, tid = threadIdx.x;
    long src = (long)seq[row] * DD;
    long dst = (long)row * DD;
    float a = emb[src + tid], b = emb[src + tid + 256];
    x[dst + tid] = a;              x[dst + tid + 256] = b;
    xh[dst + tid] = (_Float16)a;   xh[dst + tid + 256] = (_Float16)b;
}

// qh[i] = f16( q16[i] + rw[i % D] )
__global__ __launch_bounds__(256) void make_qhat(
    const _Float16* __restrict__ q, const float* __restrict__ rw,
    _Float16* __restrict__ qh, long n)
{
    long i = (long)blockIdx.x * 256 + threadIdx.x;
    if (i < n) qh[i] = (_Float16)((float)q[i] + rw[i & (DD - 1)]);
}

// ---------------------------------------------------------------------------
// score[r,c] = (AC[r,c] + [c<=r]*(Bm[r,c+S-1-r] + rb[c+S-1-r])) * SCALE,
// masked -> softmax over c. Writes P (f32, in place over AC) + f16 mirror.
// ---------------------------------------------------------------------------
__global__ __launch_bounds__(256) void attn_softmax(
    const float* __restrict__ AC, const float* __restrict__ Bm,
    const float* __restrict__ rb, const unsigned char* __restrict__ mask,
    float* __restrict__ P, _Float16* __restrict__ P16)
{
    int r = blockIdx.x;
    int z = blockIdx.y;          // b*H + h
    int b = z >> 3;
    int h = z & 7;
    int tid = threadIdx.x;
    long rowbase = (long)z * SS * SS + (long)r * SS;
    bool okr = mask[b * SS + r] != 0;

    float vals[2];
    #pragma unroll
    for (int e = 0; e < 2; ++e) {
        int c = tid + (e << 8);
        float v = AC[rowbase + c];
        if (c <= r) {
            int jj = c + SS - 1 - r;          // rel-shift gather
            v += Bm[rowbase + jj] + rb[h * MK + jj];
        }
        v *= SCALE_F;
        bool okc = mask[b * SS + c] != 0;
        vals[e] = (okr && okc) ? v : NEG_F;
    }

    __shared__ float red[256];
    red[tid] = fmaxf(vals[0], vals[1]);
    __syncthreads();
    for (int s = 128; s > 0; s >>= 1) {
        if (tid < s) red[tid] = fmaxf(red[tid], red[tid + s]);
        __syncthreads();
    }
    float rowmax = red[0];
    __syncthreads();
    float e0 = expf(vals[0] - rowmax), e1 = expf(vals[1] - rowmax);
    red[tid] = e0 + e1;
    __syncthreads();
    for (int s = 128; s > 0; s >>= 1) {
        if (tid < s) red[tid] += red[tid + s];
        __syncthreads();
    }
    float inv = 1.0f / red[0];
    float p0 = e0 * inv, p1 = e1 * inv;
    P[rowbase + tid]         = p0;
    P[rowbase + tid + 256]   = p1;
    P16[rowbase + tid]       = (_Float16)p0;
    P16[rowbase + tid + 256] = (_Float16)p1;
}

// out = LN(x + t)*g + beta  (f32, in-place safe) + f16 mirror
__global__ __launch_bounds__(256) void residual_ln(
    const float* __restrict__ x, const float* __restrict__ t,
    const float* __restrict__ g, const float* __restrict__ beta,
    float* __restrict__ out, _Float16* __restrict__ outh)
{
    long base = (long)blockIdx.x * DD;
    int tid = threadIdx.x;
    float y0 = x[base + tid]       + t[base + tid];
    float y1 = x[base + tid + 256] + t[base + tid + 256];
    __shared__ float rs[256], rq[256];
    rs[tid] = y0 + y1;
    rq[tid] = y0 * y0 + y1 * y1;
    __syncthreads();
    for (int s = 128; s > 0; s >>= 1) {
        if (tid < s) { rs[tid] += rs[tid + s]; rq[tid] += rq[tid + s]; }
        __syncthreads();
    }
    float mean = rs[0] * (1.0f / DD);
    float var  = rq[0] * (1.0f / DD) - mean * mean;
    float rstd = rsqrtf(var + 1e-5f);
    float o0 = (y0 - mean) * rstd * g[tid]       + beta[tid];
    float o1 = (y1 - mean) * rstd * g[tid + 256] + beta[tid + 256];
    out[base + tid]        = o0;
    out[base + tid + 256]  = o1;
    outh[base + tid]       = (_Float16)o0;
    outh[base + tid + 256] = (_Float16)o1;
}

// out[(b*T + t)*S + s, :] = x[b*S + s, :]
__global__ __launch_bounds__(256) void copy_out(
    const float* __restrict__ x, float* __restrict__ out, int t)
{
    int row = blockIdx.x;
    int b = row >> 9, s = row & 511;
    long dst = (((long)b * TT + t) * SS + s) * DD;
    long src = (long)row * DD;
    out[dst + threadIdx.x]       = x[src + threadIdx.x];
    out[dst + threadIdx.x + 256] = x[src + threadIdx.x + 256];
}

// ---------------------------------------------------------------------------
extern "C" void kernel_launch(void* const* d_in, const int* in_sizes, int n_in,
                              void* d_out, int out_size, void* d_ws, size_t ws_size,
                              hipStream_t stream)
{
    (void)in_sizes; (void)n_in; (void)out_size; (void)ws_size;
    const int*           seq      = (const int*)d_in[0];
    const unsigned char* mask     = (const unsigned char*)d_in[1];
    const float*         embed    = (const float*)d_in[2];
    const float*         r_emb    = (const float*)d_in[3];
    const float*         r_w_bias = (const float*)d_in[4];
    const float*         r_bias   = (const float*)d_in[5];
    const float*         wq       = (const float*)d_in[6];
    const float*         wk       = (const float*)d_in[7];
    const float*         wv       = (const float*)d_in[8];
    const float*         wo       = (const float*)d_in[9];
    const float*         ff_w1    = (const float*)d_in[10];
    const float*         ff_b1    = (const float*)d_in[11];
    const float*         ff_w2    = (const float*)d_in[12];
    const float*         ff_b2    = (const float*)d_in[13];
    const float*         ln1_g    = (const float*)d_in[14];
    const float*         ln1_b    = (const float*)d_in[15];
    const float*         ln2_g    = (const float*)d_in[16];
    const float*         ln2_b    = (const float*)d_in[17];
    float* out = (float*)d_out;

    const long XN = (long)BB * SS * DD;        // activation elems
    const long PN = (long)BB * HH * SS * SS;   // score elems
    const long FN = (long)BB * SS * FFF;
    const int  NW = TT * NLAYERS;              // 16 weight matrices per family

    char* p = (char*)d_ws;
    auto alloc = [&](long bytes) { void* r = p; p += (bytes + 255) & ~255L; return r; };
    float*    xbuf = (float*)alloc(XN * 4);
    float*    tmp  = (float*)alloc(XN * 4);
    float*    ACb  = (float*)alloc(PN * 4);
    float*    Bmb  = (float*)alloc(PN * 4);
    _Float16* xh   = (_Float16*)alloc(XN * 2);
    _Float16* q16  = (_Float16*)alloc(XN * 2);
    _Float16* qh16 = (_Float16*)alloc(XN * 2);
    _Float16* k16  = (_Float16*)alloc(XN * 2);
    _Float16* v16  = (_Float16*)alloc(XN * 2);
    _Float16* vt16 = (_Float16*)alloc(XN * 2);
    _Float16* P16  = (_Float16*)alloc(PN * 2);
    _Float16* av16 = qh16;                        // alias: qh16 dead after AC gemm
    _Float16* ffh16= P16;                         // alias: P16 dead after AV gemm (FN<PN)
    _Float16* wqT  = (_Float16*)alloc((long)NW * DD * DD * 2);
    _Float16* wkT  = (_Float16*)alloc((long)NW * DD * DD * 2);
    _Float16* wvT  = (_Float16*)alloc((long)NW * DD * DD * 2);
    _Float16* woT  = (_Float16*)alloc((long)NW * DD * DD * 2);
    _Float16* w1T  = (_Float16*)alloc((long)NW * DD * FFF * 2);
    _Float16* w2T  = (_Float16*)alloc((long)NW * DD * FFF * 2);
    _Float16* re16 = (_Float16*)alloc((long)NW * HH * SS * DHH * 2);

    // ---- one-time weight prep: f32 -> f16, transposed to (N,K) ----
    dim3 tDD(DD/32, DD/32, NW);
    trans_f32_f16<<<tDD, 256, 0, stream>>>(wq, (long)DD*DD, wqT, (long)DD*DD, DD, DD);
    trans_f32_f16<<<tDD, 256, 0, stream>>>(wk, (long)DD*DD, wkT, (long)DD*DD, DD, DD);
    trans_f32_f16<<<tDD, 256, 0, stream>>>(wv, (long)DD*DD, wvT, (long)DD*DD, DD, DD);
    trans_f32_f16<<<tDD, 256, 0, stream>>>(wo, (long)DD*DD, woT, (long)DD*DD, DD, DD);
    trans_f32_f16<<<dim3(FFF/32, DD/32, NW), 256, 0, stream>>>(
        ff_w1, (long)DD*FFF, w1T, (long)FFF*DD, FFF, DD);     // w1T[n∈FF][k∈D]
    trans_f32_f16<<<dim3(DD/32, FFF/32, NW), 256, 0, stream>>>(
        ff_w2, (long)FFF*DD, w2T, (long)DD*FFF, DD, FFF);     // w2T[n∈D][k∈FF]
    conv_re<<<dim3(SS*DHH/256, NW*HH), 256, 0, stream>>>(r_emb, re16);

    const dim3 gM(DD / 64, (BB * SS) / 64, 1);
    const dim3 gF1(FFF / 64, (BB * SS) / 64, 1);
    const dim3 gAC(SS / 64, SS / 64, BB * HH);
    const dim3 gAV(1, SS / 64, BB * HH);

    for (int t = 0; t < TT; ++t) {
        embed_gather<<<BB * SS, 256, 0, stream>>>(
            seq + (long)t * BB * SS, embed + (long)t * VOC * DD, xbuf, xh);

        for (int l = 0; l < NLAYERS; ++l) {
            const long wi = (long)t * NLAYERS + l;

            // q, k, v projections (f16 out only)
            wmma_gemm_f16<<<gM, 128, 0, stream>>>(xh, 0, 0, DD, wqT + wi*DD*DD, 0, 0, DD,
                nullptr, q16, 0, 0, DD, DD, 1, nullptr, 0);
            wmma_gemm_f16<<<gM, 128, 0, stream>>>(xh, 0, 0, DD, wkT + wi*DD*DD, 0, 0, DD,
                nullptr, k16, 0, 0, DD, DD, 1, nullptr, 0);
            wmma_gemm_f16<<<gM, 128, 0, stream>>>(xh, 0, 0, DD, wvT + wi*DD*DD, 0, 0, DD,
                nullptr, v16, 0, 0, DD, DD, 1, nullptr, 0);

            make_qhat<<<(int)((XN + 255) / 256), 256, 0, stream>>>(
                q16, r_w_bias + wi * DD, qh16, XN);

            // v -> per-(b,h) transposed (DH, S) for the AV gemm
            trans_f16_f16<<<dim3(DHH/32, SS/32, BB*HH), 256, 0, stream>>>(
                v16, (long)SS*DD, (long)DHH, HH, vt16, (long)DHH*SS, DD, SS);

            // AC = (q + r_w)_h @ k_h^T
            wmma_gemm_f16<<<gAC, 128, 0, stream>>>(
                qh16, (long)SS*DD, (long)DHH, DD,
                k16,  (long)SS*DD, (long)DHH, DD,
                ACb, nullptr, (long)HH*SS*SS, (long)SS*SS, SS,
                DHH, HH, nullptr, 0);
            // Bm = q_h @ re_h^T
            wmma_gemm_f16<<<gAC, 128, 0, stream>>>(
                q16, (long)SS*DD, (long)DHH, DD,
                re16 + wi * HH * (long)SS * DHH, 0, (long)SS*DHH, DHH,
                Bmb, nullptr, (long)HH*SS*SS, (long)SS*SS, SS,
                DHH, HH, nullptr, 0);

            attn_softmax<<<dim3(SS, BB * HH), 256, 0, stream>>>(
                ACb, Bmb, r_bias + wi * HH * (long)MK + (MK - SS),
                mask + (long)t * BB * SS, ACb, P16);

            // av = P @ v_h  (B = vt16 in (N=dh, K=s) layout)
            wmma_gemm_f16<<<gAV, 128, 0, stream>>>(
                P16,  (long)HH*SS*SS, (long)SS*SS, SS,
                vt16, (long)HH*DHH*SS, (long)DHH*SS, SS,
                nullptr, av16, (long)SS*DD, (long)DHH, DD,
                SS, HH, nullptr, 0);

            // o = av @ wo ; x = LN(x + o)
            wmma_gemm_f16<<<gM, 128, 0, stream>>>(av16, 0, 0, DD, woT + wi*DD*DD, 0, 0, DD,
                tmp, nullptr, 0, 0, DD, DD, 1, nullptr, 0);
            residual_ln<<<BB * SS, 256, 0, stream>>>(
                xbuf, tmp, ln1_g + wi * DD, ln1_b + wi * DD, xbuf, xh);

            // h = gelu(x@w1+b1) ; y = h@w2+b2 ; x = LN(x + y)
            wmma_gemm_f16<<<gF1, 128, 0, stream>>>(xh, 0, 0, DD,
                w1T + wi * (long)FFF * DD, 0, 0, DD,
                nullptr, ffh16, 0, 0, FFF, DD, 1, ff_b1 + wi * FFF, 1);
            wmma_gemm_f16<<<gM, 128, 0, stream>>>(ffh16, 0, 0, FFF,
                w2T + wi * (long)DD * FFF, 0, 0, FFF,
                tmp, nullptr, 0, 0, DD, FFF, 1, ff_b2 + wi * DD, 0);
            residual_ln<<<BB * SS, 256, 0, stream>>>(
                xbuf, tmp, ln2_g + wi * DD, ln2_b + wi * DD, xbuf, xh);
        }
        copy_out<<<BB * SS, 256, 0, stream>>>(xbuf, out, t);
    }
}
